// GCNPolicy_52398601011806
// MI455X (gfx1250) — matrix operand order repaired and checked by reading, not compiled, not run on hardware
//
#include <hip/hip_runtime.h>
#include <hip/hip_bf16.h>
#include <stdint.h>

typedef __attribute__((ext_vector_type(16))) __bf16 v16bf;
typedef __attribute__((ext_vector_type(8)))  float  v8f;
typedef __attribute__((__vector_size__(16))) int    vi4;   // builtin's b128 pointee

#define NN 100000
#define EE 1600000
#define BBATCH 64
#define FNDIM 32
#define GDIM 16
#define HDIM 128
#define ABUD 5
#define TMAX 20

#define AS1 __attribute__((address_space(1)))
#define AS3 __attribute__((address_space(3)))

// Use gfx1250 async global->LDS copies when the toolchain exposes them;
// otherwise fall back to a plain cooperative copy (still ds_store/ds_load).
#if defined(__has_builtin)
#  if __has_builtin(__builtin_amdgcn_global_load_async_to_lds_b128) && \
      __has_builtin(__builtin_amdgcn_s_wait_asynccnt)
#    define USE_ASYNC_LDS 1
#  endif
#endif
#ifndef USE_ASYNC_LDS
#  define USE_ASYNC_LDS 0
#endif

// ---------------------------------------------------------------------------
// Cooperative weight-tile staging: global (pre-packed bf16 tiles) -> LDS.
// One async b128 per chunk; waits ASYNCcnt==0 then workgroup barrier.
// ---------------------------------------------------------------------------
static __device__ __forceinline__ void stage_tiles(const uint4* __restrict__ Bp,
                                                   uint4* sB, int nch, int tid) {
  for (int i = tid; i < nch; i += 256) {
#if USE_ASYNC_LDS
    __builtin_amdgcn_global_load_async_to_lds_b128(
        (AS1 vi4*)(Bp + i), (AS3 vi4*)(sB + i), 0, 0);
#else
    sB[i] = Bp[i];
#endif
  }
#if USE_ASYNC_LDS
  __builtin_amdgcn_s_wait_asynccnt(0);
#endif
  __syncthreads();
}

// ---------------------------------------------------------------------------
// WMMA tile loaders
// ---------------------------------------------------------------------------
// A (16x32 bf16) from row-major f32 activations, ISA 7.12.2 layout:
// lanes 0-15: row M=lane, elems 0..7 -> K=kBase+0..7, elems 8..15 -> K=kBase+16..23
// lanes16-31: row M=lane-16, elems 0..7 -> K=kBase+8..15, elems 8..15 -> K=kBase+24..31
static __device__ __forceinline__ v16bf loadA_bf16(const float* __restrict__ A,
                                                   int row, int ldA, int kBase, int lane) {
  int m    = lane & 15;
  int koff = (lane & 16) ? 8 : 0;
  const float* p = A + (size_t)(row + m) * ldA + kBase + koff;
  float4 f0 = *(const float4*)(p + 0);
  float4 f1 = *(const float4*)(p + 4);
  float4 f2 = *(const float4*)(p + 16);
  float4 f3 = *(const float4*)(p + 20);
  v16bf a;
  a[0]=(__bf16)f0.x;  a[1]=(__bf16)f0.y;  a[2]=(__bf16)f0.z;  a[3]=(__bf16)f0.w;
  a[4]=(__bf16)f1.x;  a[5]=(__bf16)f1.y;  a[6]=(__bf16)f1.z;  a[7]=(__bf16)f1.w;
  a[8]=(__bf16)f2.x;  a[9]=(__bf16)f2.y;  a[10]=(__bf16)f2.z; a[11]=(__bf16)f2.w;
  a[12]=(__bf16)f3.x; a[13]=(__bf16)f3.y; a[14]=(__bf16)f3.z; a[15]=(__bf16)f3.w;
  return a;
}

union BU { uint4 u[2]; v16bf v; };

// B (32x16 bf16) from LDS-staged lane-major tile store: 2 x ds_load_b128.
static __device__ __forceinline__ v16bf loadB_lds(const uint4* sB, int tile, int lane) {
  BU t;
  const uint4* p = sB + (size_t)tile * 64 + lane * 2;
  t.u[0] = p[0];
  t.u[1] = p[1];
  return t.v;
}

// ---------------------------------------------------------------------------
// Weight packing: W[K x 128] f32 row-major -> bf16 WMMA-B tiles.
// Tile t = kstep*8 + ntile. Element i of lane l maps to
// K = kstep*32 + (l<16 ? 0 : 8) + (i&7) + (i>=8 ? 16 : 0), col = ntile*16 + (l&15).
// ---------------------------------------------------------------------------
__global__ void k_pack_w(const float* __restrict__ W, unsigned short* __restrict__ out, int K) {
  int idx = blockIdx.x * blockDim.x + threadIdx.x;
  int total = K * HDIM;
  if (idx >= total) return;
  int tile = idx >> 9;          // 512 bf16 per tile
  int rem  = idx & 511;
  int lane = rem >> 4;
  int i    = rem & 15;
  int kstep = tile >> 3, ntile = tile & 7;
  int k   = kstep * 32 + ((lane & 16) ? 8 : 0) + (i & 7) + ((i & 8) ? 16 : 0);
  int col = ntile * 16 + (lane & 15);
  __bf16 b = (__bf16)W[(size_t)k * HDIM + col];
  out[idx] = __builtin_bit_cast(unsigned short, b);
}

// ---------------------------------------------------------------------------
// Generic node GEMM: out[N,128] = act(A[N,K] @ W + bias). One wave = 16 rows.
// Weights staged once per block in LDS.
// ---------------------------------------------------------------------------
template<int K, bool RELU, bool BIAS>
__global__ void __launch_bounds__(256)
k_gemm(const float* __restrict__ A, int ldA,
       const uint4* __restrict__ Bp,
       const float* __restrict__ bias,
       float* __restrict__ out) {
  constexpr int NCH = (K / 32) * 8 * 64;   // uint4 chunks of packed tiles
  __shared__ uint4 sB[NCH];
  stage_tiles(Bp, sB, NCH, threadIdx.x);

  int wave = threadIdx.x >> 5;
  int lane = threadIdx.x & 31;
  int tile = blockIdx.x * (blockDim.x >> 5) + wave;
  if (tile * 16 < NN) {
    int row = tile * 16;
    v8f acc[8] = {};
#pragma unroll
    for (int ks = 0; ks < K / 32; ks++) {
      v16bf a = loadA_bf16(A, row, ldA, ks * 32, lane);
      if (ks + 1 < K / 32)  // warm next A kstep (global_prefetch_b8)
        __builtin_prefetch(A + (size_t)(row + (lane & 15)) * ldA + (ks + 1) * 32, 0, 0);
#pragma unroll
      for (int nt = 0; nt < 8; nt++) {
        v16bf b = loadB_lds(sB, ks * 8 + nt, lane);
        acc[nt] = __builtin_amdgcn_wmma_f32_16x16x32_bf16(
            false, a, false, b, (short)0, acc[nt], false, false);
      }
    }
    int g = lane >> 4;
    int cl = lane & 15;
#pragma unroll
    for (int nt = 0; nt < 8; nt++) {
      int col = nt * 16 + cl;
      float bv = BIAS ? bias[col] : 0.0f;
#pragma unroll
      for (int r = 0; r < 8; r++) {
        int node = row + r + 8 * g;
        float v = acc[nt][r] + bv;
        if (RELU) v = fmaxf(v, 0.0f);
        out[(size_t)node * HDIM + col] = v;
      }
    }
  }
}

// ---------------------------------------------------------------------------
// Fused source/dest head: logits[a,n] = relu(combined[n,:]@W1[a]+b1[a]) @ W2[a] + b2[a]
// Layer-2 reduction done in registers + __shfl_xor (wave32).
// grid.y: 0..4 = source action a, 5..9 = dest action a.
// ---------------------------------------------------------------------------
__global__ void __launch_bounds__(256)
k_heads(const float* __restrict__ combined,
        const uint4* __restrict__ Bp_s, const uint4* __restrict__ Bp_d,
        const float* __restrict__ s_b1, const float* __restrict__ d_b1,
        const float* __restrict__ s_W2, const float* __restrict__ d_W2,
        const float* __restrict__ s_b2, const float* __restrict__ d_b2,
        float* __restrict__ out_s, float* __restrict__ out_d) {
  __shared__ uint4 sB[4096];               // 64 tiles = 64KB (K=256)
  int hd = blockIdx.y;
  int head = hd / ABUD, a = hd % ABUD;
  const uint4* Bp = (head == 0 ? Bp_s : Bp_d) + (size_t)a * 64 * 64;
  stage_tiles(Bp, sB, 4096, threadIdx.x);

  int wave = threadIdx.x >> 5;
  int lane = threadIdx.x & 31;
  int tile = blockIdx.x * (blockDim.x >> 5) + wave;
  if (tile * 16 >= NN) return;             // after barrier: safe

  const float* b1 = (head == 0 ? s_b1 : d_b1) + a * HDIM;
  const float* W2 = (head == 0 ? s_W2 : d_W2) + a * HDIM;
  float b2        = (head == 0 ? s_b2 : d_b2)[a];
  float* out      = (head == 0 ? out_s : out_d) + (size_t)a * NN;

  int row = tile * 16;
  v8f acc[8] = {};
#pragma unroll
  for (int ks = 0; ks < 8; ks++) {                // K = 256
    v16bf av = loadA_bf16(combined, row, 2 * HDIM, ks * 32, lane);
#pragma unroll
    for (int nt = 0; nt < 8; nt++) {
      v16bf bv = loadB_lds(sB, ks * 8 + nt, lane);
      acc[nt] = __builtin_amdgcn_wmma_f32_16x16x32_bf16(
          false, av, false, bv, (short)0, acc[nt], false, false);
    }
  }
  int cl = lane & 15, g = lane >> 4;
  float part[8] = {0, 0, 0, 0, 0, 0, 0, 0};
#pragma unroll
  for (int nt = 0; nt < 8; nt++) {
    int col = nt * 16 + cl;
    float b1v = b1[col];
    float w2v = W2[col];
#pragma unroll
    for (int r = 0; r < 8; r++) {
      float hh = fmaxf(acc[nt][r] + b1v, 0.0f);
      part[r] += hh * w2v;
    }
  }
#pragma unroll
  for (int r = 0; r < 8; r++) {
    float p = part[r];
    p += __shfl_xor(p, 8, 32);
    p += __shfl_xor(p, 4, 32);
    p += __shfl_xor(p, 2, 32);
    p += __shfl_xor(p, 1, 32);
    if (cl == 0) out[row + r + 8 * g] = p + b2;
  }
}

// ---------------------------------------------------------------------------
// Graph / elementwise helpers
// ---------------------------------------------------------------------------
__global__ void k_zero(float* __restrict__ p, int n) {
  int i = blockIdx.x * blockDim.x + threadIdx.x;
  if (i < n) p[i] = 0.0f;
}

__global__ void k_deg(const int* __restrict__ dst, float* __restrict__ deg) {
  int e = blockIdx.x * blockDim.x + threadIdx.x;
  if (e < EE) atomicAdd(&deg[dst[e]], 1.0f);
}

__global__ void k_dinv(float* __restrict__ deg) {
  int n = blockIdx.x * blockDim.x + threadIdx.x;
  if (n < NN) deg[n] = rsqrtf(deg[n] + 1.0f);   // +1 self loop; always > 0
}

__global__ void k_norm(const int* __restrict__ src, const int* __restrict__ dst,
                       const float* __restrict__ dinv, float* __restrict__ norm) {
  int e = blockIdx.x * blockDim.x + threadIdx.x;
  if (e < EE) norm[e] = dinv[src[e]] * dinv[dst[e]];
}

// agg[dst] += xw[src] * norm ; 32 threads/edge, float4 each
__global__ void k_edge_agg(const int* __restrict__ src, const int* __restrict__ dst,
                           const float* __restrict__ norm, const float* __restrict__ xw,
                           float* __restrict__ agg) {
  long long idx = (long long)blockIdx.x * blockDim.x + threadIdx.x;
  int e = (int)(idx >> 5);
  if (e >= EE) return;
  int c = ((int)idx & 31) * 4;
  int s = src[e], d = dst[e];
  float nm = norm[e];
  float4 v = *(const float4*)(xw + (size_t)s * HDIM + c);
  float* ap = agg + (size_t)d * HDIM + c;
  atomicAdd(ap + 0, v.x * nm);
  atomicAdd(ap + 1, v.y * nm);
  atomicAdd(ap + 2, v.z * nm);
  atomicAdd(ap + 3, v.w * nm);
}

// h = relu((agg + self_loop + conv_b) * bn_scale + bn_b) [+ residual]
__global__ void k_post(const float* __restrict__ agg, const float* __restrict__ xw,
                       const float* __restrict__ dinv, const float* __restrict__ conv_b,
                       const float* __restrict__ bn_g, const float* __restrict__ bn_b,
                       float* __restrict__ h, int layer) {
  int i = blockIdx.x * blockDim.x + threadIdx.x;
  if (i >= NN * HDIM) return;
  int n = i >> 7, c = i & 127;
  float di = dinv[n];
  float v = agg[i] + xw[i] * di * di + conv_b[c];
  float scale = bn_g[c] * rsqrtf(1.0f + 1e-5f);
  v = fmaxf(v * scale + bn_b[c], 0.0f);
  if (layer > 0) v += h[i];
  h[i] = v;
}

__global__ void k_pool(const float* __restrict__ h, const int* __restrict__ batch,
                       float* __restrict__ pool, float* __restrict__ cnt) {
  int i = blockIdx.x * blockDim.x + threadIdx.x;
  if (i >= NN * HDIM) return;
  int n = i >> 7, c = i & 127;
  atomicAdd(&pool[batch[n] * HDIM + c], h[i]);
  if (c == 0) atomicAdd(&cnt[batch[n]], 1.0f);
}

__global__ void k_pool_div(float* __restrict__ pool, const float* __restrict__ cnt) {
  int i = blockIdx.x * blockDim.x + threadIdx.x;
  if (i >= BBATCH * HDIM) return;
  pool[i] /= fmaxf(cnt[i >> 7], 1.0f);
}

__global__ void k_gemb(const float* __restrict__ gf, const float* __restrict__ gp_W,
                       const float* __restrict__ gp_b, float* __restrict__ gemb) {
  int i = blockIdx.x * blockDim.x + threadIdx.x;
  if (i >= BBATCH * HDIM) return;
  int b = i >> 7, hc = i & 127;
  float s = gp_b[hc];
  for (int g = 0; g < GDIM; g++) s += gf[b * GDIM + g] * gp_W[g * HDIM + hc];
  gemb[i] = fmaxf(s, 0.0f);
}

__global__ void k_combined(const float* __restrict__ h, const float* __restrict__ gemb,
                           const int* __restrict__ batch, float* __restrict__ comb) {
  int i = blockIdx.x * blockDim.x + threadIdx.x;
  if (i >= NN * HDIM) return;
  int n = i >> 7, c = i & 127;
  comb[(size_t)n * 256 + c] = h[i];
  comb[(size_t)n * 256 + 128 + c] = gemb[batch[n] * HDIM + c];
}

// ----- tiny value head (B=64, scalar is fine) -----
__global__ void k_v1(const float* __restrict__ pool, const float* __restrict__ gemb,
                     const float* __restrict__ W1, const float* __restrict__ b1,
                     float* __restrict__ v1) {
  int i = blockIdx.x * blockDim.x + threadIdx.x;
  if (i >= BBATCH * HDIM) return;
  int b = i >> 7, hc = i & 127;
  float s = b1[hc];
  for (int c = 0; c < HDIM; c++) s += pool[b * HDIM + c] * W1[c * HDIM + hc];
  for (int c = 0; c < HDIM; c++) s += gemb[b * HDIM + c] * W1[(HDIM + c) * HDIM + hc];
  v1[i] = fmaxf(s, 0.0f);
}

__global__ void k_v2(const float* __restrict__ v1, const float* __restrict__ W2,
                     const float* __restrict__ b2, float* __restrict__ v2) {
  int i = blockIdx.x * blockDim.x + threadIdx.x;
  if (i >= BBATCH * 64) return;
  int b = i >> 6, hc = i & 63;
  float s = b2[hc];
  for (int c = 0; c < HDIM; c++) s += v1[b * HDIM + c] * W2[c * 64 + hc];
  v2[i] = fmaxf(s, 0.0f);
}

__global__ void k_v3(const float* __restrict__ v2, const float* __restrict__ W3,
                     const float* __restrict__ b3, float* __restrict__ out) {
  int b = threadIdx.x;
  if (b >= BBATCH) return;
  float s = b3[0];
  for (int c = 0; c < 64; c++) s += v2[b * 64 + c] * W3[c];
  out[b] = s;
}

// ----- troops head (tiny) -----
__global__ void k_troops1(const float* __restrict__ gemb, const float* __restrict__ tW1,
                          const float* __restrict__ tb1, float* __restrict__ th) {
  int i = blockIdx.x * blockDim.x + threadIdx.x;
  if (i >= ABUD * BBATCH * HDIM) return;
  int a = i / (BBATCH * HDIM);
  int r = i % (BBATCH * HDIM);
  int b = r >> 7, hc = r & 127;
  float s = tb1[a * HDIM + hc];
  for (int g = 0; g < HDIM; g++)
    s += gemb[b * HDIM + g] * tW1[(size_t)a * HDIM * HDIM + g * HDIM + hc];
  th[i] = fmaxf(s, 0.0f);
}

__global__ void k_troops2(const float* __restrict__ th, const float* __restrict__ tW2,
                          const float* __restrict__ tb2, float* __restrict__ out) {
  int i = blockIdx.x * blockDim.x + threadIdx.x;
  if (i >= ABUD * BBATCH * TMAX) return;
  int a = i / (BBATCH * TMAX);
  int r = i % (BBATCH * TMAX);
  int b = r / TMAX, t = r % TMAX;
  float s = tb2[a * TMAX + t];
  for (int hc = 0; hc < HDIM; hc++)
    s += th[(size_t)(a * BBATCH + b) * HDIM + hc] * tW2[(size_t)a * HDIM * TMAX + hc * TMAX + t];
  out[i] = s;
}

// ---------------------------------------------------------------------------
extern "C" void kernel_launch(void* const* d_in, const int* in_sizes, int n_in,
                              void* d_out, int out_size, void* d_ws, size_t ws_size,
                              hipStream_t stream) {
  const float* x    = (const float*)d_in[0];
  const int*   ei   = (const int*)d_in[1];
  const int*   batch= (const int*)d_in[2];
  const float* gf   = (const float*)d_in[3];
  const float* in_W = (const float*)d_in[4];
  const float* in_b = (const float*)d_in[5];
  const float* conv_W = (const float*)d_in[6];
  const float* conv_b = (const float*)d_in[7];
  const float* bn_g = (const float*)d_in[8];
  const float* bn_b = (const float*)d_in[9];
  const float* gp_W = (const float*)d_in[10];
  const float* gp_b = (const float*)d_in[11];
  const float* v_W1 = (const float*)d_in[12];
  const float* v_b1 = (const float*)d_in[13];
  const float* v_W2 = (const float*)d_in[14];
  const float* v_b2 = (const float*)d_in[15];
  const float* v_W3 = (const float*)d_in[16];
  const float* v_b3 = (const float*)d_in[17];
  const float* s_W1 = (const float*)d_in[18];
  const float* s_b1 = (const float*)d_in[19];
  const float* s_W2 = (const float*)d_in[20];
  const float* s_b2 = (const float*)d_in[21];
  const float* d_W1 = (const float*)d_in[22];
  const float* d_b1 = (const float*)d_in[23];
  const float* d_W2 = (const float*)d_in[24];
  const float* d_b2 = (const float*)d_in[25];
  const float* t_W1 = (const float*)d_in[26];
  const float* t_b1 = (const float*)d_in[27];
  const float* t_W2 = (const float*)d_in[28];
  const float* t_b2 = (const float*)d_in[29];

  const int* src = ei;
  const int* dst = ei + EE;
  float* out = (float*)d_out;          // [src 500000][dst 500000][troops 6400][value 64]

  // ---- workspace layout (f32 elements) ----
  float* ws = (float*)d_ws;
  size_t off = 0;
  float* dinv = ws + off; off += 100096;
  float* norm = ws + off; off += EE;
  float* h    = ws + off; off += (size_t)NN * HDIM;
  float* xw   = ws + off; off += (size_t)NN * HDIM;   // aliased by `combined`
  float* agg  = ws + off; off += (size_t)NN * HDIM;   // (xw+agg = 25.6M f32 = combined)
  float* comb = xw;
  float* pool = ws + off; off += BBATCH * HDIM;
  float* cnt  = ws + off; off += 128;
  float* gemb = ws + off; off += BBATCH * HDIM;
  float* v1   = ws + off; off += BBATCH * HDIM;
  float* v2   = ws + off; off += BBATCH * 64;
  float* th   = ws + off; off += ABUD * BBATCH * HDIM;
  off = (off + 3) & ~(size_t)3;                        // 16B align for uint4 loads
  unsigned short* pk_in   = (unsigned short*)(ws + off);
  unsigned short* pk_conv = pk_in + 8 * 512;           // 3 * 32 tiles
  unsigned short* pk_s    = pk_conv + 3 * 32 * 512;    // 5 * 64 tiles
  unsigned short* pk_d    = pk_s + 5 * 64 * 512;       // 5 * 64 tiles

  const int TPB = 256;
  const int tiles = NN / 16;                  // 6250
  const int gblocks = (tiles + 7) / 8;        // 8 waves / block
  const int nh = NN * HDIM;                   // 12.8M

  // ---- degree / norm ----
  k_zero<<<(NN + TPB - 1) / TPB, TPB, 0, stream>>>(dinv, NN);
  k_deg<<<(EE + TPB - 1) / TPB, TPB, 0, stream>>>(dst, dinv);
  k_dinv<<<(NN + TPB - 1) / TPB, TPB, 0, stream>>>(dinv);
  k_norm<<<(EE + TPB - 1) / TPB, TPB, 0, stream>>>(src, dst, dinv, norm);

  // ---- pack weights to WMMA-B tiles (bf16) ----
  k_pack_w<<<(32 * HDIM + TPB - 1) / TPB, TPB, 0, stream>>>(in_W, pk_in, 32);
  for (int l = 0; l < 3; l++)
    k_pack_w<<<(128 * HDIM + TPB - 1) / TPB, TPB, 0, stream>>>(
        conv_W + (size_t)l * HDIM * HDIM, pk_conv + (size_t)l * 32 * 512, 128);
  for (int a = 0; a < ABUD; a++) {
    k_pack_w<<<(256 * HDIM + TPB - 1) / TPB, TPB, 0, stream>>>(
        s_W1 + (size_t)a * 256 * HDIM, pk_s + (size_t)a * 64 * 512, 256);
    k_pack_w<<<(256 * HDIM + TPB - 1) / TPB, TPB, 0, stream>>>(
        d_W1 + (size_t)a * 256 * HDIM, pk_d + (size_t)a * 64 * 512, 256);
  }

  // ---- input linear (WMMA) ----
  k_gemm<32, true, true><<<gblocks, TPB, 0, stream>>>(x, FNDIM, (const uint4*)pk_in, in_b, h);

  // ---- GCN layers ----
  for (int l = 0; l < 3; l++) {
    k_gemm<128, false, false><<<gblocks, TPB, 0, stream>>>(
        h, HDIM, (const uint4*)(pk_conv + (size_t)l * 32 * 512), nullptr, xw);
    k_zero<<<(nh + TPB - 1) / TPB, TPB, 0, stream>>>(agg, nh);
    k_edge_agg<<<(int)(((long long)EE * 32 + TPB - 1) / TPB), TPB, 0, stream>>>(
        src, dst, norm, xw, agg);
    k_post<<<(nh + TPB - 1) / TPB, TPB, 0, stream>>>(
        agg, xw, dinv, conv_b + l * HDIM, bn_g + l * HDIM, bn_b + l * HDIM, h, l);
  }

  // ---- pooling + global embedding ----
  k_zero<<<(BBATCH * HDIM + 128 + TPB - 1) / TPB, TPB, 0, stream>>>(pool, BBATCH * HDIM + 128);
  k_pool<<<(nh + TPB - 1) / TPB, TPB, 0, stream>>>(h, batch, pool, cnt);
  k_pool_div<<<(BBATCH * HDIM + TPB - 1) / TPB, TPB, 0, stream>>>(pool, cnt);
  k_gemb<<<(BBATCH * HDIM + TPB - 1) / TPB, TPB, 0, stream>>>(gf, gp_W, gp_b, gemb);

  // ---- value head ----
  k_v1<<<(BBATCH * HDIM + TPB - 1) / TPB, TPB, 0, stream>>>(pool, gemb, v_W1, v_b1, v1);
  k_v2<<<(BBATCH * 64 + TPB - 1) / TPB, TPB, 0, stream>>>(v1, v_W2, v_b2, v2);
  k_v3<<<1, 64, 0, stream>>>(v2, v_W3, v_b3, out + 1006400);

  // ---- combined features + fused source/dest heads (WMMA) ----
  k_combined<<<(nh + TPB - 1) / TPB, TPB, 0, stream>>>(h, gemb, batch, comb);
  dim3 hg(gblocks, 2 * ABUD);
  k_heads<<<hg, TPB, 0, stream>>>(comb, (const uint4*)pk_s, (const uint4*)pk_d,
                                  s_b1, d_b1, s_W2, d_W2, s_b2, d_b2,
                                  out, out + 500000);

  // ---- troops head ----
  k_troops1<<<(ABUD * BBATCH * HDIM + TPB - 1) / TPB, TPB, 0, stream>>>(gemb, t_W1, t_b1, th);
  k_troops2<<<(ABUD * BBATCH * TMAX + TPB - 1) / TPB, TPB, 0, stream>>>(th, t_W2, t_b2, out + 1000000);
}
